// GraphAttentionEmbedding_63496796504581
// MI455X (gfx1250) — compile-verified, matrix operand order checked by reading, not run on hardware
//
#include <hip/hip_runtime.h>
#include <hip/hip_bf16.h>

typedef __bf16 bf16;
typedef __attribute__((ext_vector_type(4)))  bf16  bf16x4;
typedef __attribute__((ext_vector_type(8)))  bf16  bf16x8;
typedef __attribute__((ext_vector_type(16))) bf16  bf16x16;
typedef __attribute__((ext_vector_type(8)))  float f32x8;

__device__ __forceinline__ bf16x16 cat8(bf16x8 lo, bf16x8 hi) {
  return __builtin_shufflevector(lo, hi, 0,1,2,3,4,5,6,7,8,9,10,11,12,13,14,15);
}
__device__ __forceinline__ f32x8 wmma_bf16(bf16x16 a, bf16x16 b, f32x8 c) {
  return __builtin_amdgcn_wmma_f32_16x16x32_bf16(false, a, false, b, (short)0, c,
                                                 false, false);
}
// order-preserving float -> uint encoding for atomicMax-based segment max
__device__ __forceinline__ unsigned enc_f32(float f) {
  int i = __float_as_int(f);
  return (i >= 0) ? ((unsigned)i | 0x80000000u) : ~(unsigned)i;
}
__device__ __forceinline__ float dec_f32(unsigned u) {
  int i = (u & 0x80000000u) ? (int)(u & 0x7FFFFFFFu) : ~(int)u;
  return __int_as_float(i);
}

__global__ __launch_bounds__(256) void init_kernel(unsigned* menc, float* denom, int n) {
  int g = blockIdx.x * 256 + threadIdx.x;
  if (g < n) { menc[g] = 0u; denom[g] = 0.0f; }
}

// Repack W (row-major [K][128] f32) into per-fragment bf16 B-operand layout.
// Fragment f = ks*8+nb; lane holds col n = nb*16+(lane&15), K pattern per ISA:
// elems 0..7 -> K = ks*32 + (lane>>4)*8 + j ; elems 8..15 -> +16.
__global__ __launch_bounds__(256) void prepack_kernel(
    const float* __restrict__ Wq, const float* __restrict__ Wk,
    const float* __restrict__ Wv, const float* __restrict__ Wskip,
    const float* __restrict__ We, bf16* __restrict__ pack) {
  int g = blockIdx.x * 256 + threadIdx.x;
  int frag = g >> 5, lane = g & 31;
  const float* W; int fl;
  if (frag < 32)       { W = Wq;    fl = frag;       }
  else if (frag < 64)  { W = Wk;    fl = frag - 32;  }
  else if (frag < 96)  { W = Wv;    fl = frag - 64;  }
  else if (frag < 128) { W = Wskip; fl = frag - 96;  }
  else                 { W = We;    fl = frag - 128; }
  int ks = fl >> 3, nb = fl & 7;
  int n = nb * 16 + (lane & 15), khalf = lane >> 4;
  bf16x16 v;
  #pragma unroll
  for (int j = 0; j < 16; ++j) {
    int k = ks * 32 + (j < 8 ? khalf * 8 + j : 16 + khalf * 8 + (j - 8));
    v[j] = (bf16)W[k * 128 + n];
  }
  *(bf16x16*)((char*)pack + (size_t)frag * 1024 + lane * 32) = v;
}

// q_all/k_all/v_all = x@W+b, out = x@W_skip+b_skip.  128-row tiles, bf16 WMMA.
__global__ __launch_bounds__(256) void node_gemm_kernel(
    const float* __restrict__ x, const bf16* __restrict__ pack,
    const float* __restrict__ b_q, const float* __restrict__ b_k,
    const float* __restrict__ b_v, const float* __restrict__ b_skip,
    float* __restrict__ q_all, float* __restrict__ k_all,
    float* __restrict__ v_all, float* __restrict__ out, int N) {
  const int SX = 144;  // padded bf16 row stride (288B, 16B-aligned)
  __shared__ __align__(16) bf16 Xs[128 * 144];
  int tid = threadIdx.x;
  int row0 = blockIdx.x * 128;
  bool full = (row0 + 128 <= N);  // block-uniform: no bounds checks needed
  {
    int r = tid >> 1, half = tid & 1;
    int gr = row0 + r;
    bf16* Xrow = Xs + r * SX + half * 64;
    if (full || gr < N) {
      const float4* x4 = (const float4*)(x + (size_t)gr * 128 + half * 64);
      #pragma unroll
      for (int j = 0; j < 16; ++j) {
        float4 v = x4[j];
        Xrow[j*4+0] = (bf16)v.x; Xrow[j*4+1] = (bf16)v.y;
        Xrow[j*4+2] = (bf16)v.z; Xrow[j*4+3] = (bf16)v.w;
      }
    } else {
      #pragma unroll
      for (int j = 0; j < 64; ++j) Xrow[j] = (bf16)0.0f;
    }
  }
  __syncthreads();
  int lane = tid & 31, wv = tid >> 5;
  int mbase = wv * 16;
  int khalf = lane >> 4;
  const bf16* Abase = Xs + (mbase + (lane & 15)) * SX + khalf * 8;
  float* outs[4] = {q_all, k_all, v_all, out};
  const float* biases[4] = {b_q, b_k, b_v, b_skip};
  #pragma unroll
  for (int wm = 0; wm < 4; ++wm) {
    const bf16* mp = pack + (size_t)wm * 32 * 512;
    float* op = outs[wm];
    const float* bb = biases[wm];
    #pragma unroll
    for (int nb = 0; nb < 8; ++nb) {
      f32x8 acc = {};
      #pragma unroll
      for (int ks = 0; ks < 4; ++ks) {
        bf16x8 alo = *(const bf16x8*)(Abase + ks * 32);
        bf16x8 ahi = *(const bf16x8*)(Abase + ks * 32 + 16);
        const bf16* bp = mp + (ks * 8 + nb) * 512 + lane * 16;
        bf16x8 blo = *(const bf16x8*)bp;
        bf16x8 bhi = *(const bf16x8*)(bp + 8);
        acc = wmma_bf16(cat8(alo, ahi), cat8(blo, bhi), acc);
      }
      int col = nb * 16 + (lane & 15);
      float bias = bb[col];
      int rb = mbase + (lane >> 4) * 8;
      float* orow = op + (size_t)(row0 + rb) * 128 + col;
      if (full) {
        #pragma unroll
        for (int rr = 0; rr < 8; ++rr) orow[(size_t)rr * 128] = acc[rr] + bias;
      } else {
        #pragma unroll
        for (int rr = 0; rr < 8; ++rr)
          if (row0 + rb + rr < N) orow[(size_t)rr * 128] = acc[rr] + bias;
      }
    }
  }
}

// Per 64-edge tile: build edge_attr (msg | cos time enc) in LDS, WMMA e-tile,
// stage e through LDS (buffer reuse), store bf16 e, logits + scatter-max.
__global__ __launch_bounds__(256) void edge_kernel(
    const float* __restrict__ last_update, const float* __restrict__ tarr,
    const float* __restrict__ msg, const float* __restrict__ w_time,
    const float* __restrict__ b_time, const int* __restrict__ ei,
    const bf16* __restrict__ we_pack, const float* __restrict__ q_all,
    const float* __restrict__ k_all, bf16* __restrict__ e_all,
    float* __restrict__ alpha_ws, unsigned* __restrict__ menc, int E) {
  const int SA = 264;  // bf16 stride of edge_attr tile (528B)
  const int SE = 132;  // f32 stride of e tile (same bytes, reused buffer)
  __shared__ __align__(16) unsigned char smem[64 * 264 * 2];
  __shared__ int   s_src[64], s_dst[64];
  __shared__ float s_rel[64], s_wt[128], s_bt[128];
  bf16*  As = (bf16*)smem;
  float* Ef = (float*)smem;
  int tid = threadIdx.x;
  int eb = blockIdx.x * 64;
  if (tid < 64) {
    int ge = eb + tid;
    int src = ei[ge];
    s_src[tid] = src;
    s_dst[tid] = ei[E + ge];
    s_rel[tid] = last_update[src] - tarr[ge];
  } else if (tid < 192) {
    int d = tid - 64;
    s_wt[d] = w_time[d];
    s_bt[d] = b_time[d];
  }
  __syncthreads();
  {  // edge_attr: rows=edges, cols 0..127 = msg, 128..255 = cos(rel*w+b)
    int r = tid >> 2, qq = tid & 3;
    int ge = eb + r;
    bf16* Arow = As + r * SA;
    if (qq < 2) {
      const float4* m4 = (const float4*)(msg + (size_t)ge * 128 + qq * 64);
      if (ge + 64 < E)
        __builtin_prefetch(msg + (size_t)(ge + 64) * 128 + qq * 64, 0, 0);
      #pragma unroll
      for (int j = 0; j < 16; ++j) {
        float4 v = m4[j];
        int c = qq * 64 + j * 4;
        Arow[c+0] = (bf16)v.x; Arow[c+1] = (bf16)v.y;
        Arow[c+2] = (bf16)v.z; Arow[c+3] = (bf16)v.w;
      }
    } else {
      float rel = s_rel[r];
      int d0 = (qq - 2) * 64;
      #pragma unroll 8
      for (int j = 0; j < 64; ++j) {
        int d = d0 + j;
        Arow[128 + d] = (bf16)__cosf(fmaf(rel, s_wt[d], s_bt[d]));
      }
    }
  }
  __syncthreads();
  // e[64x128] = edge_attr[64x256] @ We[256x128]; wave -> (4 M-sub x 2 N-half)
  int lane = tid & 31, wv = tid >> 5;
  int mbase = (wv & 3) * 16;
  int nh = wv >> 2;
  int khalf = lane >> 4;
  const bf16* Abase = As + (mbase + (lane & 15)) * SA + khalf * 8;
  f32x8 acc[4] = {};
  #pragma unroll
  for (int ks = 0; ks < 8; ++ks) {
    bf16x8 alo = *(const bf16x8*)(Abase + ks * 32);
    bf16x8 ahi = *(const bf16x8*)(Abase + ks * 32 + 16);
    bf16x16 a = cat8(alo, ahi);
    #pragma unroll
    for (int i = 0; i < 4; ++i) {
      int nb = nh * 4 + i;
      const bf16* bp = we_pack + (ks * 8 + nb) * 512 + lane * 16;
      bf16x8 blo = *(const bf16x8*)bp;
      bf16x8 bhi = *(const bf16x8*)(bp + 8);
      acc[i] = wmma_bf16(a, cat8(blo, bhi), acc[i]);
    }
  }
  __syncthreads();  // all edge_attr reads done -> reuse smem as f32 e tile
  {
    int rb = mbase + (lane >> 4) * 8;
    #pragma unroll
    for (int i = 0; i < 4; ++i) {
      int col = (nh * 4 + i) * 16 + (lane & 15);
      #pragma unroll
      for (int rr = 0; rr < 8; ++rr) Ef[(rb + rr) * SE + col] = acc[i][rr];
    }
  }
  __syncthreads();
  {  // store e as bf16 (halves HBM traffic of the largest intermediate)
    int r = tid >> 2, qq = tid & 3;
    const float* erow = Ef + r * SE + qq * 32;
    bf16* gout = e_all + (size_t)(eb + r) * 128 + qq * 32;
    #pragma unroll
    for (int blk = 0; blk < 4; ++blk) {
      bf16x8 v;
      #pragma unroll
      for (int j = 0; j < 8; ++j) v[j] = (bf16)erow[blk * 8 + j];
      *(bf16x8*)(gout + blk * 8) = v;
    }
  }
  if (tid < 128) {  // alpha[e,h] = q[dst]·(k[src]+e) / sqrt(64); scatter-max
    int el = tid >> 1, h = tid & 1;
    int src = s_src[el], dst = s_dst[el];
    const float4* qp = (const float4*)(q_all + (size_t)dst * 128 + h * 64);
    const float4* kp = (const float4*)(k_all + (size_t)src * 128 + h * 64);
    const float* ep = Ef + el * SE + h * 64;
    float s = 0.0f;
    #pragma unroll
    for (int j = 0; j < 16; ++j) {
      float4 qv = qp[j], kv = kp[j];
      s = fmaf(qv.x, kv.x + ep[j*4+0], s);
      s = fmaf(qv.y, kv.y + ep[j*4+1], s);
      s = fmaf(qv.z, kv.z + ep[j*4+2], s);
      s = fmaf(qv.w, kv.w + ep[j*4+3], s);
    }
    s *= 0.125f;
    alpha_ws[(size_t)(eb + el) * 2 + h] = s;
    atomicMax(menc + (size_t)dst * 2 + h, enc_f32(s));
  }
}

// one thread per edge, both heads: ea = exp(alpha - m[dst]); denom += ea
__global__ __launch_bounds__(256) void exp_kernel(
    const float* __restrict__ alpha_ws, const int* __restrict__ ei,
    const unsigned* __restrict__ menc, float* __restrict__ ea,
    float* __restrict__ denom, int E) {
  int e = blockIdx.x * 256 + threadIdx.x;
  if (e >= E) return;
  int dst = ei[E + e];
  float2 a = *(const float2*)(alpha_ws + (size_t)e * 2);
  uint2 mu = *(const uint2*)(menc + (size_t)dst * 2);
  float2 v;
  v.x = __expf(a.x - dec_f32(mu.x));
  v.y = __expf(a.y - dec_f32(mu.y));
  *(float2*)(ea + (size_t)e * 2) = v;
  atomicAdd(denom + (size_t)dst * 2 + 0, v.x);
  atomicAdd(denom + (size_t)dst * 2 + 1, v.y);
}

// 4 feature elements per thread: vectorized loads, dst-uniform coalesced atomics
__global__ __launch_bounds__(256) void scatter_kernel(
    const int* __restrict__ ei, const float* __restrict__ ea,
    const float* __restrict__ denom, const float* __restrict__ v_all,
    const bf16* __restrict__ e_all, float* __restrict__ out, int E) {
  long g = (long)blockIdx.x * 256 + threadIdx.x;
  if (g >= (long)E * 32) return;
  int e = (int)(g >> 5), d = (int)(g & 31) * 4, h = d >> 6;
  int src = ei[e], dst = ei[E + e];
  float attn = ea[(size_t)e * 2 + h] / (denom[(size_t)dst * 2 + h] + 1e-16f);
  float4 v = *(const float4*)(v_all + (size_t)src * 128 + d);
  bf16x4 ev = *(const bf16x4*)(e_all + (size_t)e * 128 + d);
  float* od = out + (size_t)dst * 128 + d;
  atomicAdd(od + 0, (v.x + (float)ev[0]) * attn);
  atomicAdd(od + 1, (v.y + (float)ev[1]) * attn);
  atomicAdd(od + 2, (v.z + (float)ev[2]) * attn);
  atomicAdd(od + 3, (v.w + (float)ev[3]) * attn);
}

extern "C" void kernel_launch(void* const* d_in, const int* in_sizes, int n_in,
                              void* d_out, int out_size, void* d_ws, size_t ws_size,
                              hipStream_t stream) {
  const float* x        = (const float*)d_in[0];
  const float* last_upd = (const float*)d_in[1];
  const float* tarr     = (const float*)d_in[2];
  const float* msg      = (const float*)d_in[3];
  const float* w_time   = (const float*)d_in[4];
  const float* b_time   = (const float*)d_in[5];
  const float* W_q      = (const float*)d_in[6];
  const float* b_q      = (const float*)d_in[7];
  const float* W_k      = (const float*)d_in[8];
  const float* b_k      = (const float*)d_in[9];
  const float* W_v      = (const float*)d_in[10];
  const float* b_v      = (const float*)d_in[11];
  const float* W_e      = (const float*)d_in[12];
  const float* W_skip   = (const float*)d_in[13];
  const float* b_skip   = (const float*)d_in[14];
  const int*   ei       = (const int*)d_in[15];
  int N = in_sizes[0] / 128;
  int E = in_sizes[2];
  float* out = (float*)d_out;

  char* w = (char*)d_ws;
  bf16* pack     = (bf16*)w;      w += 192 * 1024;             // packed weights
  float* q_all   = (float*)w;     w += (size_t)N * 128 * 4;
  float* k_all   = (float*)w;     w += (size_t)N * 128 * 4;
  float* v_all   = (float*)w;     w += (size_t)N * 128 * 4;
  bf16* e_all    = (bf16*)w;      w += (size_t)E * 128 * 2;
  float* alpha   = (float*)w;     w += (size_t)E * 2 * 4;
  float* ea      = (float*)w;     w += (size_t)E * 2 * 4;
  unsigned* menc = (unsigned*)w;  w += (size_t)N * 2 * 4;
  float* denom   = (float*)w;

  init_kernel<<<(2 * N + 255) / 256, 256, 0, stream>>>(menc, denom, 2 * N);
  prepack_kernel<<<24, 256, 0, stream>>>(W_q, W_k, W_v, W_skip, W_e, pack);
  node_gemm_kernel<<<(N + 127) / 128, 256, 0, stream>>>(
      x, pack, b_q, b_k, b_v, b_skip, q_all, k_all, v_all, out, N);
  edge_kernel<<<E / 64, 256, 0, stream>>>(
      last_upd, tarr, msg, w_time, b_time, ei, pack + 128 * 512,
      q_all, k_all, e_all, alpha, menc, E);
  exp_kernel<<<(E + 255) / 256, 256, 0, stream>>>(alpha, ei, menc, ea, denom, E);
  long total = (long)E * 32;
  scatter_kernel<<<(unsigned)((total + 255) / 256), 256, 0, stream>>>(
      ei, ea, denom, v_all, e_all, out, E);
}
